// HybridSparseTransformerLayer_42674795053798
// MI455X (gfx1250) — matrix operand order repaired and checked by reading, not compiled, not run on hardware
//
#include <hip/hip_runtime.h>
#include <hip/hip_bf16.h>

// ---------------------------------------------------------------------------
// Types / helpers
// ---------------------------------------------------------------------------
typedef __bf16 bf16;
typedef __attribute__((ext_vector_type(16))) __bf16 v16bf;
typedef __attribute__((ext_vector_type(8)))  float  v8f;
typedef __attribute__((ext_vector_type(4)))  int    v4i;

// Native converts: gfx1250 has bf16 converts; let the compiler emit them.
__device__ __forceinline__ bf16 f2bf(float f) { return (bf16)f; }
__device__ __forceinline__ float bf2f(bf16 h) { return (float)h; }

// Async global->LDS staging (CDNA5). Probe confirmed the builtin exists and
// takes (int4 AS1* gsrc, int4 AS3* ldsdst, imm offset, imm cpol).
// AS pointers are produced via integer casts: AS1 shares the flat bit
// pattern; AS3 = low 32 bits of the flat address (ISA 10.2 aperture rule).
#if defined(__has_builtin)
#if __has_builtin(__builtin_amdgcn_global_load_async_to_lds_b128)
#define HAVE_ASYNC_LDS 1
#endif
#if __has_builtin(__builtin_amdgcn_s_wait_asynccnt)
#define HAVE_WAIT_ASYNC_BUILTIN 1
#endif
#endif

#ifdef HAVE_ASYNC_LDS
typedef __attribute__((address_space(1))) v4i as1_v4i;
typedef __attribute__((address_space(3))) v4i as3_v4i;
__device__ __forceinline__ void async_copy16(const void* gsrc, void* ldsdst) {
    __builtin_amdgcn_global_load_async_to_lds_b128(
        (as1_v4i*)(unsigned long long)gsrc,
        (as3_v4i*)(unsigned int)(unsigned long long)ldsdst,
        0, 0);
}
__device__ __forceinline__ void wait_async0() {
#ifdef HAVE_WAIT_ASYNC_BUILTIN
    __builtin_amdgcn_s_wait_asynccnt(0);
#else
    asm volatile("s_wait_asynccnt 0" ::: "memory");
#endif
}
#endif

// Problem constants
#define CB    4
#define CS    2048
#define CD    1024
#define CH    16
#define CFFN  4096
#define CFRES 2048
#define CE    8
#define CCAP  2048
#define NTOK  (CB * CS)

// ---------------------------------------------------------------------------
// Kernel 1: fused LN1 + RoPE (full-D rope, half = 512) and LN3, bf16 outputs
// ---------------------------------------------------------------------------
__global__ __launch_bounds__(256) void ln1_rope_ln3_kernel(
    const float* __restrict__ x,
    const float* __restrict__ g1, const float* __restrict__ b1,
    const float* __restrict__ g3, const float* __restrict__ b3,
    bf16* __restrict__ hout, bf16* __restrict__ sout)
{
    int tok = blockIdx.x;          // b*S + s
    int s   = tok % CS;
    const float* xp = x + (long)tok * CD;
    int tid = threadIdx.x;

    float vals[4], sum = 0.f, sq = 0.f;
#pragma unroll
    for (int i = 0; i < 4; i++) {
        float v = xp[tid + i * 256];
        vals[i] = v; sum += v; sq += v * v;
    }
    for (int d = 1; d < 32; d <<= 1) { sum += __shfl_xor(sum, d, 32); sq += __shfl_xor(sq, d, 32); }
    __shared__ float red[2][8];
    int wave = tid >> 5, lane = tid & 31;
    if (lane == 0) { red[0][wave] = sum; red[1][wave] = sq; }
    __syncthreads();
    if (wave == 0) {
        float a = (lane < 8) ? red[0][lane] : 0.f;
        float b = (lane < 8) ? red[1][lane] : 0.f;
        for (int d = 1; d < 8; d <<= 1) { a += __shfl_xor(a, d, 32); b += __shfl_xor(b, d, 32); }
        if (lane == 0) { red[0][0] = a; red[1][0] = b; }
    }
    __syncthreads();
    float mean = red[0][0] * (1.0f / CD);
    float var  = red[1][0] * (1.0f / CD) - mean * mean;
    float rs   = rsqrtf(var + 1e-5f);

    // LN3 output
#pragma unroll
    for (int i = 0; i < 4; i++) {
        int d = tid + i * 256;
        float nv = (vals[i] - mean) * rs;
        sout[(long)tok * CD + d] = f2bf(nv * g3[d] + b3[d]);
    }
    // LN1 then RoPE: pairs (d, d+512); each thread owns both halves
    float y[4];
#pragma unroll
    for (int i = 0; i < 4; i++) {
        int d = tid + i * 256;
        y[i] = (vals[i] - mean) * rs * g1[d] + b1[d];
    }
#pragma unroll
    for (int i = 0; i < 2; i++) {
        int d1 = tid + i * 256;
        float inv = __expf(-(float)d1 * (9.210340371976184f / 512.0f)); // 10000^(-d1/512)
        float ang = (float)s * inv;
        float sn, cs;
        __sincosf(ang, &sn, &cs);
        float x1v = y[i], x2v = y[i + 2];
        hout[(long)tok * CD + d1]       = f2bf(x1v * cs - x2v * sn);
        hout[(long)tok * CD + d1 + 512] = f2bf(x1v * sn + x2v * cs);
    }
}

// ---------------------------------------------------------------------------
// Kernel 2: plain LayerNorm (fp32 in -> bf16 out)  (used for ln2 on x1)
// ---------------------------------------------------------------------------
__global__ __launch_bounds__(256) void ln_bf16_kernel(
    const float* __restrict__ x,
    const float* __restrict__ g, const float* __restrict__ b,
    bf16* __restrict__ out)
{
    int tok = blockIdx.x;
    const float* xp = x + (long)tok * CD;
    int tid = threadIdx.x;
    float vals[4], sum = 0.f, sq = 0.f;
#pragma unroll
    for (int i = 0; i < 4; i++) {
        float v = xp[tid + i * 256];
        vals[i] = v; sum += v; sq += v * v;
    }
    for (int d = 1; d < 32; d <<= 1) { sum += __shfl_xor(sum, d, 32); sq += __shfl_xor(sq, d, 32); }
    __shared__ float red[2][8];
    int wave = tid >> 5, lane = tid & 31;
    if (lane == 0) { red[0][wave] = sum; red[1][wave] = sq; }
    __syncthreads();
    if (wave == 0) {
        float a = (lane < 8) ? red[0][lane] : 0.f;
        float c = (lane < 8) ? red[1][lane] : 0.f;
        for (int d = 1; d < 8; d <<= 1) { a += __shfl_xor(a, d, 32); c += __shfl_xor(c, d, 32); }
        if (lane == 0) { red[0][0] = a; red[1][0] = c; }
    }
    __syncthreads();
    float mean = red[0][0] * (1.0f / CD);
    float var  = red[1][0] * (1.0f / CD) - mean * mean;
    float rs   = rsqrtf(var + 1e-5f);
#pragma unroll
    for (int i = 0; i < 4; i++) {
        int d = tid + i * 256;
        out[(long)tok * CD + d] = f2bf((vals[i] - mean) * rs * g[d] + b[d]);
    }
}

// ---------------------------------------------------------------------------
// Kernel 3: generic WMMA GEMM, C = A(bf16, MxK) * B(fp32 weights) + epilogue
//   b_nk = 1 : B stored NxK (out = A @ W^T)      -- torch-style weights
//   b_nk = 0 : B stored KxN (out = A @ W)        -- expert weights
// Tile: 128x64, K-step 32, 8 waves (4 along M x 2 along N), each wave 32x32.
// A tile staged via GLOBAL_LOAD_ASYNC_TO_LDS_B128 (ASYNCcnt), B tile
// converted fp32->bf16 in-flight (overlaps the async A copy).
// Epilogues: 0 fp32 +bias | 1 fp32 +bias+residual | 2 bf16 +bias
//            3 bf16 gelu(acc) | 4 scatter-atomicAdd(acc*gate) via idx list
// ---------------------------------------------------------------------------
#define BM 128
#define BN 64
#define BK 32
#define LDAP 40   // padded LDS row (bf16 elems): 80 B, keeps 16B alignment
#define LDBP 40

__global__ __launch_bounds__(256) void gemm_bf16_wmma(
    const bf16*  __restrict__ A, long aBatch,
    const float* __restrict__ Bm, long bBatch,
    const float* __restrict__ bias,
    float* outF, bf16* outB, long oBatch,
    const float* __restrict__ residual,
    const int*   __restrict__ idxList,
    const float* __restrict__ gate, long idxBatch,
    int M, int N, int K, int epi, int b_nk)
{
    __shared__ bf16 As[BM * LDAP];
    __shared__ bf16 Bs[BN * LDBP];

    int z = blockIdx.z;
    const bf16*  Ap = A  + (long)z * aBatch;
    const float* Bp = Bm + (long)z * bBatch;
    const int*   idxp  = idxList ? idxList + (long)z * idxBatch : nullptr;
    const float* gatep = gate    ? gate    + (long)z * idxBatch : nullptr;

    int m0 = blockIdx.y * BM;
    int n0 = blockIdx.x * BN;
    int tid  = threadIdx.x;
    int lane = tid & 31, wave = tid >> 5;
    int wm = (wave & 3) * 32, wn = (wave >> 2) * 32;

    v8f acc[2][2] = {};

    for (int kk = 0; kk < K; kk += BK) {
        // ---- stage A tile (128 x 32 bf16): 32 bytes / thread ----
        {
            int r = tid >> 1, c = (tid & 1) * 16;
            const bf16* src = Ap + (long)(m0 + r) * K + kk + c;
            if (kk + BK < K)
                __builtin_prefetch(src + BK, 0, 3);     // -> global_prefetch_b8
#ifdef HAVE_ASYNC_LDS
            async_copy16(src,     &As[r * LDAP + c]);
            async_copy16(src + 8, &As[r * LDAP + c + 8]);
#else
            uint4 d0 = *(const uint4*)(src);
            uint4 d1 = *(const uint4*)(src + 8);
            *(uint4*)&As[r * LDAP + c]     = d0;
            *(uint4*)&As[r * LDAP + c + 8] = d1;
#endif
        }
        // ---- stage B tile (64 x 32 bf16), convert fp32 -> bf16 ----
        if (b_nk) {                  // B is N x K
            int n = tid >> 2, c = (tid & 3) * 8;
            const float* src = Bp + (long)(n0 + n) * K + kk + c;
#pragma unroll
            for (int j = 0; j < 8; j++) Bs[n * LDBP + c + j] = f2bf(src[j]);
        } else {                     // B is K x N
            int kr = tid >> 3, c = (tid & 7) * 8;
            const float* src = Bp + (long)(kk + kr) * N + n0 + c;
#pragma unroll
            for (int j = 0; j < 8; j++) Bs[(c + j) * LDBP + kr] = f2bf(src[j]);
        }
#ifdef HAVE_ASYNC_LDS
        wait_async0();
#endif
        __syncthreads();

        // ---- fragments (ISA 7.12.2 layouts) ----
        v16bf af[2], bfrag[2];
#pragma unroll
        for (int i = 0; i < 2; i++) {
            int mb = wm + i * 16 + (lane & 15);
            int kb = (lane >> 4) << 3;
#pragma unroll
            for (int e = 0; e < 16; e++) {
                int k = kb + (e & 7) + ((e >> 3) << 4);
                af[i][e] = As[mb * LDAP + k];
            }
        }
#pragma unroll
        for (int j = 0; j < 2; j++) {
            int nb  = wn + j * 16 + (lane & 15);
            int kb2 = (lane >> 4) << 4;
#pragma unroll
            for (int e = 0; e < 16; e++) bfrag[j][e] = Bs[nb * LDBP + (kb2 >> 1) + (e & 7) + ((e >> 3) << 4)];
        }
#pragma unroll
        for (int i = 0; i < 2; i++)
#pragma unroll
            for (int j = 0; j < 2; j++)
                acc[i][j] = __builtin_amdgcn_wmma_f32_16x16x32_bf16(
                    false, af[i], false, bfrag[j], (short)0, acc[i][j], false, false);
        __syncthreads();
    }

    // ---- epilogue ----
    int rbase = (lane >> 4) << 3;
    int coll  = lane & 15;
#pragma unroll
    for (int i = 0; i < 2; i++) {
#pragma unroll
        for (int j = 0; j < 2; j++) {
            int ng = n0 + wn + j * 16 + coll;
            float bv = bias ? bias[ng] : 0.f;
#pragma unroll
            for (int r = 0; r < 8; r++) {
                int mg  = m0 + wm + i * 16 + rbase + r;
                float v = acc[i][j][r] + bv;
                long off = (long)z * oBatch + (long)mg * N + ng;
                if (epi == 0) {
                    outF[off] = v;
                } else if (epi == 1) {
                    outF[off] = v + residual[(long)mg * N + ng];
                } else if (epi == 2) {
                    outB[off] = f2bf(v);
                } else if (epi == 3) {
                    float gl = 0.5f * v * (1.0f + erff(v * 0.70710678118f));
                    outB[off] = f2bf(gl);
                } else { // 4: weighted scatter-add into d_out
                    float w  = gatep[mg];
                    int tok  = idxp[mg];
                    atomicAdd(&outF[(long)tok * N + ng], v * w);
                }
            }
        }
    }
}

// ---------------------------------------------------------------------------
// Kernel 4: flash attention, hd=64, WMMA for QK^T and P*V, online softmax.
// grid: (S/128, B*H); 8 waves/block; each wave = 16 query rows.
// ---------------------------------------------------------------------------
__global__ __launch_bounds__(256) void attn_flash_kernel(
    const bf16* __restrict__ qkv, bf16* __restrict__ o)
{
    const int TD = 3 * CD;
    int bh = blockIdx.y, b = bh / CH, h = bh % CH;
    int lane = threadIdx.x & 31, wave = threadIdx.x >> 5;
    int qrow = blockIdx.x * 128 + wave * 16;

    const bf16* qp = qkv + (long)b * CS * TD + h * 64;
    const bf16* kp = qp + CD;
    const bf16* vp = qp + 2 * CD;

    int m  = lane & 15;
    int kb = (lane >> 4) << 3;

    // Q fragments (A layout), hd split 0..31 / 32..63
    v16bf qf[2];
#pragma unroll
    for (int i = 0; i < 2; i++)
#pragma unroll
        for (int e = 0; e < 16; e++) {
            int k = i * 32 + kb + (e & 7) + ((e >> 3) << 4);
            qf[i][e] = qp[(long)(qrow + m) * TD + k];
        }

    v8f acc[4] = {};
    float mstat[8], lstat[8];
#pragma unroll
    for (int r = 0; r < 8; r++) { mstat[r] = -1e30f; lstat[r] = 0.f; }

    __shared__ float pshare[8][16][32];
    float* myp = &pshare[wave][0][0];
    const float scale = 0.125f; // 1/sqrt(64)
    int rowoff = (lane >> 4) << 3;

    for (int kt = 0; kt < CS; kt += 32) {
        // scores for 32 keys = two 16x16 tiles
        v8f sc[2] = {};
#pragma unroll
        for (int j = 0; j < 2; j++) {
            int kcol = kt + j * 16 + (lane & 15);   // key index for this lane
            int kb2  = (lane >> 4) << 4;
#pragma unroll
            for (int i = 0; i < 2; i++) {
                v16bf kf;
#pragma unroll
                for (int e = 0; e < 16; e++)
                    kf[e] = kp[(long)kcol * TD + i * 32 + kb + (e & 7) + ((e >> 3) << 4)];
                sc[j] = __builtin_amdgcn_wmma_f32_16x16x32_bf16(
                    false, qf[i], false, kf, (short)0, sc[j], false, false);
            }
        }
        // online softmax update (row reductions over 16-lane halves)
        float newm[8];
#pragma unroll
        for (int r = 0; r < 8; r++) {
            float v = fmaxf(sc[0][r], sc[1][r]) * scale;
            for (int d = 1; d < 16; d <<= 1) v = fmaxf(v, __shfl_xor(v, d, 32));
            newm[r] = fmaxf(mstat[r], v);
        }
#pragma unroll
        for (int r = 0; r < 8; r++) {
            float alpha = __expf(mstat[r] - newm[r]);
            mstat[r] = newm[r];
            float p0 = __expf(sc[0][r] * scale - newm[r]);
            float p1 = __expf(sc[1][r] * scale - newm[r]);
            sc[0][r] = p0; sc[1][r] = p1;
            float s = p0 + p1;
            for (int d = 1; d < 16; d <<= 1) s += __shfl_xor(s, d, 32);
            lstat[r] = lstat[r] * alpha + s;
#pragma unroll
            for (int t = 0; t < 4; t++) acc[t][r] *= alpha;
        }
        // C-layout -> A-layout transpose of P through LDS (same-wave, DS in order)
#pragma unroll
        for (int r = 0; r < 8; r++) {
            myp[(rowoff + r) * 32 + (lane & 15)]      = sc[0][r];
            myp[(rowoff + r) * 32 + 16 + (lane & 15)] = sc[1][r];
        }
        __builtin_amdgcn_wave_barrier();
        asm volatile("s_wait_dscnt 0" ::: "memory");
        v16bf pf;
#pragma unroll
        for (int e = 0; e < 16; e++) {
            int k = (kb << 1) + e;
            pf[e] = f2bf(myp[m * 32 + k]);
        }
        // P (16x32) x V (32 x 64)
        int kb2 = (lane >> 4) << 4;
#pragma unroll
        for (int t = 0; t < 4; t++) {
            v16bf vf;
            int n = t * 16 + (lane & 15);
#pragma unroll
            for (int e = 0; e < 16; e++)
                vf[e] = vp[(long)(kt + kb2 + e) * TD + n];
            acc[t] = __builtin_amdgcn_wmma_f32_16x16x32_bf16(
                false, pf, false, vf, (short)0, acc[t], false, false);
        }
    }
    // write o[b, s, h*64 + n]
#pragma unroll
    for (int t = 0; t < 4; t++) {
        int n = t * 16 + (lane & 15);
#pragma unroll
        for (int r = 0; r < 8; r++) {
            int srow = qrow + rowoff + r;
            float v = acc[t][r] / lstat[r];
            o[(long)b * CS * CD + (long)srow * CD + h * 64 + n] = f2bf(v);
        }
    }
}

// ---------------------------------------------------------------------------
// Kernel 5: SwiGLU gate: g[tok,j] = a * silu(b), a=h2[:, :2048], b=h2[:, 2048:]
// ---------------------------------------------------------------------------
__global__ __launch_bounds__(256) void swiglu_kernel(
    const bf16* __restrict__ h2, bf16* __restrict__ g)
{
    long i   = (long)blockIdx.x * 256 + threadIdx.x;  // over NTOK * FFN/2
    long tok = i / (CFFN / 2), j = i % (CFFN / 2);
    float a = bf2f(h2[tok * CFFN + j]);
    float b = bf2f(h2[tok * CFFN + (CFFN / 2) + j]);
    float s = b / (1.f + __expf(-b));
    g[i] = f2bf(a * s);
}

// ---------------------------------------------------------------------------
// Kernel 6: router: logits = s_in @ router_w (D x E), softmax, top-2
// one wave per token: lane&7 = expert, lane>>3 = D-stripe
// ---------------------------------------------------------------------------
__global__ __launch_bounds__(256) void router_kernel(
    const bf16* __restrict__ sin_, const float* __restrict__ rw,
    float* __restrict__ probs, int* __restrict__ top2)
{
    int tok  = blockIdx.x * 8 + (threadIdx.x >> 5);
    int lane = threadIdx.x & 31;
    int e = lane & 7, stripe = lane >> 3;
    float acc = 0.f;
    for (int d = stripe; d < CD; d += 4)
        acc += bf2f(sin_[(long)tok * CD + d]) * rw[(long)d * CE + e];
    acc += __shfl_xor(acc, 8, 32);
    acc += __shfl_xor(acc, 16, 32);
    float mx = acc;
    for (int d = 1; d < 8; d <<= 1) mx = fmaxf(mx, __shfl_xor(mx, d, 32));
    float p = __expf(acc - mx);
    float sum = p;
    for (int d = 1; d < 8; d <<= 1) sum += __shfl_xor(sum, d, 32);
    p /= sum;
    if (lane < 8) probs[(long)tok * CE + e] = p;
    float q[8];
#pragma unroll
    for (int j = 0; j < 8; j++) q[j] = __shfl(p, j, 32);
    if (lane == 0) {
        float best = -1.f, second = -1.f; int bi = 0, si = 0;
#pragma unroll
        for (int j = 0; j < 8; j++) {
            if (q[j] > best)        { second = best; si = bi; best = q[j]; bi = j; }
            else if (q[j] > second) { second = q[j]; si = j; }
        }
        top2[tok * 2] = bi; top2[tok * 2 + 1] = si;
    }
}

// ---------------------------------------------------------------------------
// Kernel 7: per-expert capacity scan (matches reference flat cumsum semantics)
// single block; phase 1 stages the 8-bit expert mask per token in LDS.
// ---------------------------------------------------------------------------
__global__ __launch_bounds__(256) void capacity_kernel(
    const int* __restrict__ top2, const float* __restrict__ probs,
    int* __restrict__ idxList, float* __restrict__ gate, int* __restrict__ cnts)
{
    __shared__ unsigned char mask[NTOK];
    for (int t = threadIdx.x; t < NTOK; t += 256) {
        int a = top2[2 * t], b = top2[2 * t + 1];
        mask[t] = (unsigned char)((1u << a) | (1u << b));
    }
    __syncthreads();
    int e = threadIdx.x;
    if (e < CE) {
        int cnt = 0;
        for (int t = 0; t < NTOK; t++) {
            if (mask[t] & (1u << e)) {
                if (cnt < CCAP) {
                    idxList[e * CCAP + cnt] = t;
                    gate[e * CCAP + cnt]    = probs[(long)t * CE + e];
                }
                cnt++;
            }
        }
        int c = cnt < CCAP ? cnt : CCAP;
        for (int i = c; i < CCAP; i++) { idxList[e * CCAP + i] = 0; gate[e * CCAP + i] = 0.f; }
        cnts[e] = c;
    }
}

// ---------------------------------------------------------------------------
// Kernel 8: gather selected tokens into padded per-expert matrices (bf16)
// grid = E*CAP blocks of 128 threads (8 bf16 / thread)
// ---------------------------------------------------------------------------
__global__ __launch_bounds__(128) void moe_gather_kernel(
    const bf16* __restrict__ sin_, const int* __restrict__ idxList,
    const int* __restrict__ cnts, bf16* __restrict__ Ain)
{
    int row = blockIdx.x;                 // e*CAP + i
    int e = row / CCAP, i = row % CCAP;
    long dst = (long)row * CD + threadIdx.x * 8;
    if (i < cnts[e]) {
        int tok = idxList[row];
        *(uint4*)&Ain[dst] = *(const uint4*)&sin_[(long)tok * CD + threadIdx.x * 8];
    } else {
        uint4 zz = {0u, 0u, 0u, 0u};
        *(uint4*)&Ain[dst] = zz;
    }
}

// ---------------------------------------------------------------------------
// Host-side launch
// ---------------------------------------------------------------------------
static void gemm(hipStream_t st, const bf16* A, long aB, const float* Bm, long bB,
                 const float* bias, float* outF, bf16* outB, long oB,
                 const float* res, const int* idx, const float* gt, long iB,
                 int M, int N, int K, int epi, int bnk, int Z)
{
    dim3 g(N / BN, M / BM, Z);
    gemm_bf16_wmma<<<g, 256, 0, st>>>(A, aB, Bm, bB, bias, outF, outB, oB,
                                      res, idx, gt, iB, M, N, K, epi, bnk);
}

extern "C" void kernel_launch(void* const* d_in, const int* in_sizes, int n_in,
                              void* d_out, int out_size, void* d_ws, size_t ws_size,
                              hipStream_t stream)
{
    (void)in_sizes; (void)n_in; (void)out_size; (void)ws_size;
    const float* x          = (const float*)d_in[0];
    const float* ln1_g      = (const float*)d_in[1];
    const float* ln1_b      = (const float*)d_in[2];
    const float* ln2_g      = (const float*)d_in[3];
    const float* ln2_b      = (const float*)d_in[4];
    const float* ln3_g      = (const float*)d_in[5];
    const float* ln3_b      = (const float*)d_in[6];
    const float* in_proj_w  = (const float*)d_in[7];
    const float* in_proj_b  = (const float*)d_in[8];
    const float* out_proj_w = (const float*)d_in[9];
    const float* out_proj_b = (const float*)d_in[10];
    const float* ff_w1      = (const float*)d_in[11];
    const float* ff_b1      = (const float*)d_in[12];
    const float* ff_w2      = (const float*)d_in[13];
    const float* ff_b2      = (const float*)d_in[14];
    const float* router_w   = (const float*)d_in[15];
    const float* expert_wi  = (const float*)d_in[16];
    const float* expert_wo  = (const float*)d_in[17];
    float* out = (float*)d_out;

    char* ws = (char*)d_ws;
    const long MB = 1 << 20;
    bf16*  slotA = (bf16*)(ws);                    // 16 MB : h / o / ln2h
    bf16*  s_in  = (bf16*)(ws + 16 * MB);          // 16 MB : persists
    bf16*  slotC = (bf16*)(ws + 32 * MB);          // 64 MB : qkv / h2 / hidden
    float* x1    = (float*)(ws + 96 * MB);         // 32 MB
    bf16*  slotE = (bf16*)(ws + 128 * MB);         // 32 MB : g / gathered Ain
    float* probs = (float*)(ws + 160 * MB);        // 256 KB
    int*   top2  = (int*)(ws + 160 * MB + 512 * 1024);
    int*   idxL  = (int*)(ws + 161 * MB);          // 64 KB
    float* gate  = (float*)(ws + 161 * MB + 256 * 1024);
    int*   cnts  = (int*)(ws + 162 * MB);

    // 1) LN1 + RoPE -> h ; LN3 -> s_in
    ln1_rope_ln3_kernel<<<NTOK, 256, 0, stream>>>(x, ln1_g, ln1_b, ln3_g, ln3_b, slotA, s_in);
    // 2) qkv = h @ in_proj_w^T + b   (bf16 out)
    gemm(stream, slotA, 0, in_proj_w, 0, in_proj_b, nullptr, slotC, 0,
         nullptr, nullptr, nullptr, 0, NTOK, 3 * CD, CD, 2, 1, 1);
    // 3) attention -> o (bf16, slotA)
    attn_flash_kernel<<<dim3(CS / 128, CB * CH), 256, 0, stream>>>(slotC, slotA);
    // 4) x1 = x + o @ out_proj_w^T + b   (fp32)
    gemm(stream, slotA, 0, out_proj_w, 0, out_proj_b, x1, nullptr, 0,
         x, nullptr, nullptr, 0, NTOK, CD, CD, 1, 1, 1);
    // 5) ln2(x1) -> slotA (bf16)
    ln_bf16_kernel<<<NTOK, 256, 0, stream>>>(x1, ln2_g, ln2_b, slotA);
    // 6) h2 = ln2h @ ff_w1^T + b1   (bf16, slotC)
    gemm(stream, slotA, 0, ff_w1, 0, ff_b1, nullptr, slotC, 0,
         nullptr, nullptr, nullptr, 0, NTOK, CFFN, CD, 2, 1, 1);
    // 7) g = a * silu(b)  (bf16, slotE)
    swiglu_kernel<<<(NTOK * (CFFN / 2)) / 256, 256, 0, stream>>>(slotC, slotE);
    // 8) dense_out = x1 + g @ ff_w2^T + b2 -> d_out (fp32, initializes output)
    gemm(stream, slotE, 0, ff_w2, 0, ff_b2, out, nullptr, 0,
         x1, nullptr, nullptr, 0, NTOK, CD, CFFN / 2, 1, 1, 1);
    // 9) router softmax + top2
    router_kernel<<<NTOK / 8, 256, 0, stream>>>(s_in, router_w, probs, top2);
    // 10) capacity-limited assignment (per-expert flat cumsum)
    capacity_kernel<<<1, 256, 0, stream>>>(top2, probs, idxL, gate, cnts);
    // 11) gather tokens per expert (padded to CAP)
    moe_gather_kernel<<<CE * CCAP, 128, 0, stream>>>(s_in, idxL, cnts, slotE);
    // 12) hidden = gelu(Ain @ expert_wi[e])   (bf16, slotC), B is KxN
    gemm(stream, slotE, (long)CCAP * CD, expert_wi, (long)CD * CFRES, nullptr,
         nullptr, slotC, (long)CCAP * CFRES, nullptr, nullptr, nullptr, 0,
         CCAP, CFRES, CD, 3, 0, CE);
    // 13) d_out[token] += gate * (hidden @ expert_wo[e])   (scatter atomics)
    gemm(stream, slotC, (long)CCAP * CFRES, expert_wo, (long)CFRES * CD, nullptr,
         out, nullptr, 0, nullptr, idxL, gate, (long)CCAP,
         CCAP, CD, CFRES, 4, 0, CE);
}
